// Attention_19361712570996
// MI455X (gfx1250) — matrix operand order repaired and checked by reading, not compiled, not run on hardware
//
#include <hip/hip_runtime.h>
#include <hip/hip_bf16.h>
#include <math.h>

typedef __attribute__((ext_vector_type(16))) __bf16 v16bf;
typedef __attribute__((ext_vector_type(8)))  __bf16 v8bf;
typedef __attribute__((ext_vector_type(4)))  __bf16 v4bf;
typedef __attribute__((ext_vector_type(8)))  float  v8f;

#define S_LEN 2048
#define HID   4096
#define NH    32
#define NKV   8
#define HD    128
#define QKV_N 6144

__device__ __forceinline__ v8f zero8() {
  v8f z = {0.f,0.f,0.f,0.f,0.f,0.f,0.f,0.f};
  return z;
}

__device__ __forceinline__ v16bf cat16(v8bf lo, v8bf hi) {
  return __builtin_shufflevector(lo, hi, 0,1,2,3,4,5,6,7,8,9,10,11,12,13,14,15);
}

// ---------------------------------------------------------------------------
// f32 -> bf16 bulk convert (4 elements / thread)
// ---------------------------------------------------------------------------
__global__ __launch_bounds__(256)
void f32_to_bf16_kernel(const float* __restrict__ in, void* __restrict__ outp) {
  __bf16* out = (__bf16*)outp;
  const size_t i = (size_t)(blockIdx.x * 256 + threadIdx.x) * 4;
  const float4 v = *(const float4*)(in + i);
  v4bf o;
  o[0] = (__bf16)v.x; o[1] = (__bf16)v.y; o[2] = (__bf16)v.z; o[3] = (__bf16)v.w;
  *(v4bf*)(out + i) = o;
}

// ---------------------------------------------------------------------------
// Tiled GEMM: C[M,N] f32 = A[M,K] (bf16, global-direct fragments)
//                        * B[K,N] (f32 -> bf16, LDS transposed, double-buffered)
// Block tile 128(M) x 256(N), 8 waves in 2x4, wave tile 64x64, K-step 32.
// ---------------------------------------------------------------------------
__global__ __launch_bounds__(256)
void gemm_wmma_kernel(const void* __restrict__ Ap, const float* __restrict__ B,
                      float* __restrict__ C, int M, int N, int K) {
  __shared__ __bf16 BsT[2][256][40];     // [buf][n][k] transposed, padded row

  const __bf16* Ab = (const __bf16*)Ap;
  const int tid   = threadIdx.x;
  const int lane  = tid & 31;
  const int wave  = tid >> 5;          // 0..7
  const int wm    = wave >> 2;         // 0..1 (M direction)
  const int wn    = wave & 3;          // 0..3 (N direction)
  const int khalf = lane >> 4;
  const int l16   = lane & 15;
  const int block_m = blockIdx.y * 128;
  const int block_n = blockIdx.x * 256;

  v8f acc[4][4];
#pragma unroll
  for (int mm = 0; mm < 4; mm++)
#pragma unroll
    for (int nn = 0; nn < 4; nn++) acc[mm][nn] = zero8();

  // thread owns tile column n = tid; loads 32 K values as 8 float4 quads
  float4 breg[8];
  auto load_breg = [&](int kk) {
#pragma unroll
    for (int p = 0; p < 8; p++) {
      const float* bp = B + (size_t)(kk + p * 4) * N + block_n + tid;
      breg[p].x = bp[0];
      breg[p].y = bp[(size_t)N];
      breg[p].z = bp[(size_t)2 * N];
      breg[p].w = bp[(size_t)3 * N];
    }
  };

  load_breg(0);
  int buf = 0;
  for (int k0 = 0; k0 < K; k0 += 32) {
    // stage staged B registers -> BsT[buf] (bf16, transposed)
#pragma unroll
    for (int p = 0; p < 8; p++) {
      v4bf o;
      o[0] = (__bf16)breg[p].x; o[1] = (__bf16)breg[p].y;
      o[2] = (__bf16)breg[p].z; o[3] = (__bf16)breg[p].w;
      *(v4bf*)&BsT[buf][tid][p * 4] = o;
    }
    __syncthreads();
    if (k0 + 32 < K) load_breg(k0 + 32);   // prefetch next tile into registers

    // B fragments: K contiguous per lane -> two ds_load_b128
    v16bf bfrag[4];
#pragma unroll
    for (int nn = 0; nn < 4; nn++) {
      const int n = wn * 64 + nn * 16 + l16;
      const v8bf blo = *(const v8bf*)&BsT[buf][n][16 * khalf];
      const v8bf bhi = *(const v8bf*)&BsT[buf][n][16 * khalf + 8];
      bfrag[nn] = cat16(blo, bhi);
    }
    // A fragments: straight from global (two b128 per lane), then 16 WMMAs
#pragma unroll
    for (int mm = 0; mm < 4; mm++) {
      const __bf16* arow = Ab + (size_t)(block_m + wm * 64 + mm * 16 + l16) * K
                              + k0 + 16 * khalf;
      const v8bf alo = *(const v8bf*)(arow);
      const v8bf ahi = *(const v8bf*)(arow + 8);
      const v16bf af = cat16(alo, ahi);
#pragma unroll
      for (int nn = 0; nn < 4; nn++)
        acc[mm][nn] = __builtin_amdgcn_wmma_f32_16x16x32_bf16(
            false, af, false, bfrag[nn], (short)0, acc[mm][nn], false, false);
    }
    buf ^= 1;
  }

  // store C (f32). D layout: VGPR r -> row r + 8*khalf, col l16
#pragma unroll
  for (int mm = 0; mm < 4; mm++)
#pragma unroll
    for (int nn = 0; nn < 4; nn++)
#pragma unroll
      for (int r = 0; r < 8; r++) {
        const int m = block_m + wm * 64 + mm * 16 + r + 8 * khalf;
        const int n = block_n + wn * 64 + nn * 16 + l16;
        C[(size_t)m * N + n] = acc[mm][nn][r];
      }
}

// ---------------------------------------------------------------------------
// RoPE + scatter: qkv f32 [S, 6144] -> Q bf16 [H][S][D], K bf16 [G][S][D],
// V^T bf16 [G][D][S]. per-position work = 32*64 + 8*64 + 8*128 = 3584 items.
// ---------------------------------------------------------------------------
__global__ __launch_bounds__(256)
void rope_scatter_kernel(const float* __restrict__ qkv,
                         const float* __restrict__ fcos, const float* __restrict__ fsin,
                         void* __restrict__ Qp, void* __restrict__ Kp, void* __restrict__ VTp) {
  __bf16* Q  = (__bf16*)Qp;
  __bf16* Kc = (__bf16*)Kp;
  __bf16* VT = (__bf16*)VTp;
  const int idx = blockIdx.x * blockDim.x + threadIdx.x;
  const int s = idx / 3584;
  const int t = idx - s * 3584;
  const float* row = qkv + (size_t)s * QKV_N;
  if (t < 2048) {                       // Q heads, RoPE pair (i, i+64)
    const int hh = t >> 6, i = t & 63;
    const float x1 = row[hh * HD + i], x2 = row[hh * HD + i + 64];
    const float c = fcos[s * 64 + i], sn = fsin[s * 64 + i];
    __bf16* dst = Q + ((size_t)hh * S_LEN + s) * HD;
    dst[i]      = (__bf16)(x1 * c - x2 * sn);
    dst[i + 64] = (__bf16)(x1 * sn + x2 * c);
  } else if (t < 2560) {                // K groups, RoPE
    const int tt = t - 2048, gg = tt >> 6, i = tt & 63;
    const float x1 = row[HID + gg * HD + i], x2 = row[HID + gg * HD + i + 64];
    const float c = fcos[s * 64 + i], sn = fsin[s * 64 + i];
    __bf16* dst = Kc + ((size_t)gg * S_LEN + s) * HD;
    dst[i]      = (__bf16)(x1 * c - x2 * sn);
    dst[i + 64] = (__bf16)(x1 * sn + x2 * c);
  } else {                              // V groups, straight cvt + transpose
    const int tt = t - 2560, gg = tt >> 7, d = tt & 127;
    const float x = row[HID + 1024 + gg * HD + d];
    VT[((size_t)gg * HD + d) * S_LEN + s] = (__bf16)x;
  }
}

// ---------------------------------------------------------------------------
// Flash attention (causal, GQA). Block = 4 waves, each wave owns a 16-row
// query tile of one head; streams key chunks of 32 with online softmax.
// ---------------------------------------------------------------------------
__global__ __launch_bounds__(128)
void flash_attn_kernel(const void* __restrict__ Qp, const void* __restrict__ Kp,
                       const void* __restrict__ VTp, void* __restrict__ ATTNp) {
  const __bf16* Q  = (const __bf16*)Qp;
  const __bf16* Kc = (const __bf16*)Kp;
  const __bf16* VT = (const __bf16*)VTp;
  __bf16* ATTN = (__bf16*)ATTNp;

  const int h    = blockIdx.y;
  const int g    = h >> 2;                 // H/KV = 4
  const int wave = threadIdx.x >> 5;
  const int lane = threadIdx.x & 31;
  const int khalf = lane >> 4, l16 = lane & 15;
  const int qb = blockIdx.x * 64 + wave * 16;

  __shared__ __bf16 Pld[4][16][34];        // per-wave P staging (C->A relayout)

  // Q fragments: 4 K-chunks of 32 over D=128
  v16bf qfrag[4];
  {
    const __bf16* qrow = Q + ((size_t)h * S_LEN + qb + l16) * HD;
#pragma unroll
    for (int c = 0; c < 4; c++)
#pragma unroll
      for (int e = 0; e < 16; e++)
        qfrag[c][e] = qrow[32 * c + 16 * khalf + e];
  }

  float mrow[8], lrow[8];
  v8f o[8];
#pragma unroll
  for (int r = 0; r < 8; r++) { mrow[r] = -INFINITY; lrow[r] = 0.f; }
#pragma unroll
  for (int c = 0; c < 8; c++) o[c] = zero8();

  const float scale = 0.08838834764831845f;   // 1/sqrt(128)
  const int jmax = (qb + 15) >> 5;

  for (int j = 0; j <= jmax; j++) {
    const int kbase = j * 32;
    v8f s0 = zero8(), s1 = zero8();
    const __bf16* kr0 = Kc + ((size_t)g * S_LEN + kbase + l16) * HD + 16 * khalf;
    const __bf16* kr1 = kr0 + 16 * HD;
#pragma unroll
    for (int c = 0; c < 4; c++) {
      const v16bf b0 = *(const v16bf*)(kr0 + 32 * c);
      const v16bf b1 = *(const v16bf*)(kr1 + 32 * c);
      s0 = __builtin_amdgcn_wmma_f32_16x16x32_bf16(false, qfrag[c], false, b0, (short)0, s0, false, false);
      s1 = __builtin_amdgcn_wmma_f32_16x16x32_bf16(false, qfrag[c], false, b1, (short)0, s1, false, false);
    }

    __bf16 (*P)[34] = Pld[wave];
#pragma unroll
    for (int r = 0; r < 8; r++) {
      const int qi = qb + r + 8 * khalf;
      float v0 = s0[r] * scale; if (kbase + l16 > qi)      v0 = -INFINITY;
      float v1 = s1[r] * scale; if (kbase + 16 + l16 > qi) v1 = -INFINITY;
      float rm = fmaxf(v0, v1);
#pragma unroll
      for (int off = 1; off < 16; off <<= 1) rm = fmaxf(rm, __shfl_xor(rm, off, 32));
      const float mnew = fmaxf(mrow[r], rm);
      const float corr = __expf(mrow[r] - mnew);
      const float p0 = __expf(v0 - mnew);
      const float p1 = __expf(v1 - mnew);
      float rs = p0 + p1;
#pragma unroll
      for (int off = 1; off < 16; off <<= 1) rs += __shfl_xor(rs, off, 32);
      lrow[r] = lrow[r] * corr + rs;
      mrow[r] = mnew;
#pragma unroll
      for (int c = 0; c < 8; c++) o[c][r] *= corr;
      P[r + 8 * khalf][l16]      = (__bf16)p0;
      P[r + 8 * khalf][16 + l16] = (__bf16)p1;
    }

    // reload P in A-fragment layout (row = l16, K = key within chunk)
    v16bf pfrag;
#pragma unroll
    for (int e = 0; e < 16; e++)
      pfrag[e] = P[l16][16 * khalf + e];

    // O += P * V  (V^T layout makes B-fragment loads contiguous)
#pragma unroll
    for (int c = 0; c < 8; c++) {
      const v16bf vfrag = *(const v16bf*)(VT + ((size_t)g * HD + 16 * c + l16) * S_LEN + kbase + 16 * khalf);
      o[c] = __builtin_amdgcn_wmma_f32_16x16x32_bf16(false, pfrag, false, vfrag, (short)0, o[c], false, false);
    }
  }

  // normalize and store attn output bf16 [S, H*D]
#pragma unroll
  for (int c = 0; c < 8; c++)
#pragma unroll
    for (int r = 0; r < 8; r++) {
      const int qi = qb + r + 8 * khalf;
      ATTN[(size_t)qi * HID + h * HD + 16 * c + l16] = (__bf16)(o[c][r] / lrow[r]);
    }
}

// ---------------------------------------------------------------------------
extern "C" void kernel_launch(void* const* d_in, const int* in_sizes, int n_in,
                              void* d_out, int out_size, void* d_ws, size_t ws_size,
                              hipStream_t stream) {
  (void)in_sizes; (void)n_in; (void)out_size; (void)ws_size;
  const float* x    = (const float*)d_in[0];
  const float* Wqkv = (const float*)d_in[1];
  const float* Wo   = (const float*)d_in[2];
  const float* fcos = (const float*)d_in[6];
  const float* fsin = (const float*)d_in[7];
  float* out = (float*)d_out;

  char* ws = (char*)d_ws;
  float* qkv = (float*)ws;                 ws += (size_t)S_LEN * QKV_N * sizeof(float);
  void* xbf  = (void*)ws;                  ws += (size_t)S_LEN * HID * 2;
  void* qb   = (void*)ws;                  ws += (size_t)NH  * S_LEN * HD * 2;
  void* kb   = (void*)ws;                  ws += (size_t)NKV * S_LEN * HD * 2;
  void* vT   = (void*)ws;                  ws += (size_t)NKV * S_LEN * HD * 2;
  void* attn = (void*)ws;

  // 0) x -> bf16
  f32_to_bf16_kernel<<<(S_LEN * HID) / (256 * 4), 256, 0, stream>>>(x, xbf);
  // 1) QKV projection: [2048,4096] x [4096,6144]
  gemm_wmma_kernel<<<dim3(QKV_N / 256, S_LEN / 128), 256, 0, stream>>>(
      xbf, Wqkv, qkv, S_LEN, QKV_N, HID);
  // 2) RoPE + bf16 scatter (Q, K, V^T)
  rope_scatter_kernel<<<(S_LEN * 3584) / 256, 256, 0, stream>>>(qkv, fcos, fsin, qb, kb, vT);
  // 3) causal GQA flash attention
  flash_attn_kernel<<<dim3(S_LEN / 64, NH), 128, 0, stream>>>(qb, kb, vT, attn);
  // 4) output projection: [2048,4096] x [4096,4096]
  gemm_wmma_kernel<<<dim3(HID / 256, S_LEN / 128), 256, 0, stream>>>(
      attn, Wo, out, S_LEN, HID, HID);
}
